// RNNEncoder_65790309040711
// MI455X (gfx1250) — matrix operand order repaired and checked by reading, not compiled, not run on hardware
//
#include <hip/hip_runtime.h>
#include <hip/hip_bf16.h>

// ---------------------------------------------------------------------------
// 2-layer ReLU RNN for MI455X (gfx1250), wave32 + WMMA bf16.
//   B=64, T=512, H=I=1024.
//   * Weights pre-swizzled once into the per-lane WMMA B-fragment layout.
//   * Hidden state kept in bf16 (row-major) -> recurrent GEMM passes are pure
//     global_load_b128 + v_wmma_f32_16x16x32_bf16 (no per-step VALU cvt).
//   * x*mask pre-scaled to bf16 once (if workspace allows) so the L0 input
//     pass is conversion-free too; fp32 fallback path kept as template.
//   * One persistent kernel: 32 blocks x 8 waves = 256 waves = 256 output
//     tiles of the [64 x 1024] layer state; atomic-counter grid barrier
//     between the two layer phases of each timestep.
// ---------------------------------------------------------------------------

typedef __attribute__((ext_vector_type(16))) __bf16 v16bf;
typedef __attribute__((ext_vector_type(8)))  __bf16 v8bf;
typedef __attribute__((ext_vector_type(8)))  float  v8f;

static constexpr int B = 64, T = 512, H = 1024;
static constexpr int HB = B * H;            // elements per hidden-state buffer
static constexpr int NBLOCKS  = 32;         // 4 row-tile groups * 8 col groups
static constexpr int NTHREADS = 256;        // 8 waves -> 8 col tiles per block

// ---- workspace layout (bytes) ----
static constexpr size_t PW_BYTES  = (size_t)H * H * 2;     // 2 MB per packed W
static constexpr size_t OFF_PW0   = 0;
static constexpr size_t OFF_PW1   = PW_BYTES;
static constexpr size_t OFF_PW2   = 2 * PW_BYTES;
static constexpr size_t OFF_PW3   = 3 * PW_BYTES;
static constexpr size_t OFF_BIAS0 = 4 * PW_BYTES;
static constexpr size_t OFF_BIAS1 = OFF_BIAS0 + 4096;
static constexpr size_t OFF_H0    = OFF_BIAS0 + 8192;            // bf16 [2][HB]
static constexpr size_t OFF_H1    = OFF_H0 + (size_t)2 * HB * 2;
static constexpr size_t OFF_CNT   = OFF_H1 + (size_t)2 * HB * 2;
static constexpr size_t OFF_XBF   = (OFF_CNT + 256 + 255) & ~(size_t)255;
static constexpr size_t XBF_BYTES = (size_t)B * T * H * 2;       // 64 MB
static constexpr size_t WS_NEED_XPRE = OFF_XBF + XBF_BYTES;
static constexpr size_t ZERO_BYTES = (size_t)4 * HB * 2 + 256;   // h0,h1,cnt

// ---------------------------------------------------------------------------
// Weight packing: B-fragment element e of lane L for (col-tile c, k-chunk kc)
// is  W[16c + L%16][32kc + 8*(L/16) + e (+8 for e>=8)]   (B[k][n] = W^T[k][n] = W[n][k]).
// Stored flat: dst[((c*32 + kc)*32 + lane)*16 + e]  -> 32B/lane contiguous.
// ---------------------------------------------------------------------------
__global__ __launch_bounds__(256) void pack_weight(const float* __restrict__ W,
                                                   __bf16* __restrict__ dst) {
  int idx  = blockIdx.x * 256 + threadIdx.x;     // 0 .. H*H-1
  int e    = idx & 15;
  int lane = (idx >> 4) & 31;
  int kc   = (idx >> 9) & 31;
  int c    = idx >> 14;
  int n    = (c << 4) + (lane & 15);
  int k    = (kc << 5) + ((lane >> 4) << 3) + (e & 7) + ((e >> 3) << 4);
  dst[idx] = (__bf16)W[n * H + k];
}

__global__ void bias_sum(const float* __restrict__ a, const float* __restrict__ b,
                         const float* __restrict__ c, const float* __restrict__ d,
                         float* __restrict__ o0, float* __restrict__ o1) {
  int i = blockIdx.x * 256 + threadIdx.x;
  if (i < H) { o0[i] = a[i] + b[i]; o1[i] = c[i] + d[i]; }
}

// x*mask -> bf16, 8 elements per thread (two float4 loads, one 16B store).
__global__ __launch_bounds__(256) void scale_x(const float* __restrict__ x,
                                               const float* __restrict__ mask,
                                               __bf16* __restrict__ xbf) {
  size_t i = ((size_t)blockIdx.x * 256 + threadIdx.x) * 8;
  float  m = mask[i >> 10];                      // i/H -> b*T + t
  float4 f0 = *(const float4*)(x + i);
  float4 f1 = *(const float4*)(x + i + 4);
  v8bf o;
  o[0] = (__bf16)(f0.x * m); o[1] = (__bf16)(f0.y * m);
  o[2] = (__bf16)(f0.z * m); o[3] = (__bf16)(f0.w * m);
  o[4] = (__bf16)(f1.x * m); o[5] = (__bf16)(f1.y * m);
  o[6] = (__bf16)(f1.z * m); o[7] = (__bf16)(f1.w * m);
  *(v8bf*)(xbf + i) = o;
}

// ---------------------------------------------------------------------------
// A-fragment (16x32 bf16): lane L holds row M = L%16; element e maps to
// K = 8*(L/16) + e (+8 for e>=8).  p points at row + k0, k0 = 32kc + 8*half.
// ---------------------------------------------------------------------------
__device__ __forceinline__ v16bf load_a_bf(const __bf16* __restrict__ p) {
  v8bf lo = *(const v8bf*)(p);
  v8bf hi = *(const v8bf*)(p + 16);
  v16bf a;
#pragma unroll
  for (int i = 0; i < 8; ++i) { a[i] = lo[i]; a[i + 8] = hi[i]; }
  return a;
}

__device__ __forceinline__ v16bf make_a_f32(const float* __restrict__ p, float s) {
  float4 f0 = *(const float4*)(p + 0);
  float4 f1 = *(const float4*)(p + 4);
  float4 f2 = *(const float4*)(p + 16);
  float4 f3 = *(const float4*)(p + 20);
  v16bf a;
  a[0]  = (__bf16)(f0.x * s); a[1]  = (__bf16)(f0.y * s);
  a[2]  = (__bf16)(f0.z * s); a[3]  = (__bf16)(f0.w * s);
  a[4]  = (__bf16)(f1.x * s); a[5]  = (__bf16)(f1.y * s);
  a[6]  = (__bf16)(f1.z * s); a[7]  = (__bf16)(f1.w * s);
  a[8]  = (__bf16)(f2.x * s); a[9]  = (__bf16)(f2.y * s);
  a[10] = (__bf16)(f2.z * s); a[11] = (__bf16)(f2.w * s);
  a[12] = (__bf16)(f3.x * s); a[13] = (__bf16)(f3.y * s);
  a[14] = (__bf16)(f3.z * s); a[15] = (__bf16)(f3.w * s);
  return a;
}

__device__ __forceinline__ v8f wmma_bf(v16bf a, v16bf b, v8f c) {
  return __builtin_amdgcn_wmma_f32_16x16x32_bf16(false, a, false, b, (short)0,
                                                 c, false, false);
}

// K=1024 as 32 WMMAs; A already bf16 row-major -> pure load+wmma.
__device__ __forceinline__ v8f kloop_bf(v8f acc, const __bf16* __restrict__ arow,
                                        const __bf16* __restrict__ pw,
                                        int c, int lane, int half) {
  const __bf16* bp = pw + (size_t)c * (32 * 512) + ((size_t)lane << 4);
  const __bf16* ap = arow + (half << 3);
#pragma unroll 4
  for (int kc = 0; kc < 32; ++kc)
    acc = wmma_bf(load_a_bf(ap + (kc << 5)),
                  *(const v16bf*)(bp + kc * 512), acc);
  return acc;
}

// fp32-A fallback (x path when workspace too small for pre-scaled x).
__device__ __forceinline__ v8f kloop_f32(v8f acc, const float* __restrict__ arow,
                                         const __bf16* __restrict__ pw,
                                         int c, int lane, int half, float s) {
  const __bf16* bp = pw + (size_t)c * (32 * 512) + ((size_t)lane << 4);
  const float*  ap = arow + (half << 3);
#pragma unroll 4
  for (int kc = 0; kc < 32; ++kc)
    acc = wmma_bf(make_a_f32(ap + (kc << 5), s),
                  *(const v16bf*)(bp + kc * 512), acc);
  return acc;
}

// Grid-wide barrier: monotonically increasing agent-scope counter.
__device__ __forceinline__ void grid_sync(unsigned* cnt, unsigned target) {
  __threadfence();
  __syncthreads();
  if (threadIdx.x == 0) {
    __hip_atomic_fetch_add(cnt, 1u, __ATOMIC_ACQ_REL, __HIP_MEMORY_SCOPE_AGENT);
    while (__hip_atomic_load(cnt, __ATOMIC_ACQUIRE, __HIP_MEMORY_SCOPE_AGENT) < target)
      __builtin_amdgcn_s_sleep(2);
  }
  __syncthreads();
  __threadfence();
}

// ---------------------------------------------------------------------------
// Persistent RNN kernel. XPRE: x*mask pre-scaled to bf16 in workspace.
// ---------------------------------------------------------------------------
template <bool XPRE>
__global__ __launch_bounds__(NTHREADS) void rnn_persistent(
    const float* __restrict__ x, const float* __restrict__ mask,
    const __bf16* __restrict__ xbf,
    const __bf16* __restrict__ pwih0, const __bf16* __restrict__ pwhh0,
    const __bf16* __restrict__ pwih1, const __bf16* __restrict__ pwhh1,
    const float* __restrict__ bias0, const float* __restrict__ bias1,
    __bf16* __restrict__ h0buf, __bf16* __restrict__ h1buf,
    unsigned* __restrict__ cnt, float* __restrict__ out) {
  const int lane    = threadIdx.x & 31;
  const int wave    = threadIdx.x >> 5;
  const int r       = blockIdx.x & 3;                  // row tile (batch/16)
  const int c       = ((blockIdx.x >> 2) << 3) + wave; // col tile (0..63)
  const int ln      = lane & 15;
  const int half    = lane >> 4;
  const int ba      = (r << 4) + ln;                   // batch row fed into A
  const int colbase = (c << 4) + ln;                   // output column (C/D)
  const float b0v = bias0[colbase];
  const float b1v = bias1[colbase];
  const float*  xrow0  = x   + (size_t)ba * T * H;
  const __bf16* xbrow0 = xbf + (size_t)ba * T * H;
  const size_t  fbase  = (size_t)B * T * H;            // h_final region
  const unsigned nb = gridDim.x;
  unsigned target = 0;

  for (int t = 0; t < T; ++t) {
    const int wsel = t & 1;        // write buffer
    const int rsel = wsel ^ 1;     // read buffer (zeroed for t==0)

    // -------- layer 0: relu(x_t*m @ Wih0^T + h0 @ Whh0^T + b0) --------
    v8f acc;
#pragma unroll
    for (int j = 0; j < 8; ++j) acc[j] = b0v;
    if (XPRE) {
      acc = kloop_bf(acc, xbrow0 + (size_t)t * H, pwih0, c, lane, half);
    } else {
      const float m = mask[ba * T + t];
      acc = kloop_f32(acc, xrow0 + (size_t)t * H, pwih0, c, lane, half, m);
    }
    acc = kloop_bf(acc, h0buf + rsel * HB + ba * H, pwhh0, c, lane, half);
    {
      __bf16* hw = h0buf + wsel * HB;
#pragma unroll
      for (int j = 0; j < 8; ++j) {
        float v = acc[j] > 0.f ? acc[j] : 0.f;
        int row = (r << 4) + (half << 3) + j;          // C row = 16r + 8*half+j
        hw[row * H + colbase] = (__bf16)v;
        if (t == T - 1) out[fbase + (size_t)row * H + colbase] = v;  // h_final[0]
      }
    }
    target += nb;
    grid_sync(cnt, target);

    // -------- layer 1: relu(h0 @ Wih1^T + h1 @ Whh1^T + b1) --------
#pragma unroll
    for (int j = 0; j < 8; ++j) acc[j] = b1v;
    acc = kloop_bf(acc, h0buf + wsel * HB + ba * H, pwih1, c, lane, half);
    acc = kloop_bf(acc, h1buf + rsel * HB + ba * H, pwhh1, c, lane, half);
    {
      __bf16* hw = h1buf + wsel * HB;
#pragma unroll
      for (int j = 0; j < 8; ++j) {
        float v = acc[j] > 0.f ? acc[j] : 0.f;
        int row = (r << 4) + (half << 3) + j;
        hw[row * H + colbase] = (__bf16)v;
        out[(size_t)row * T * H + (size_t)t * H + colbase] = v;      // outs[B,T,H]
        if (t == T - 1) out[fbase + (size_t)HB + (size_t)row * H + colbase] = v;
      }
    }
    target += nb;
    grid_sync(cnt, target);
  }
}

// ---------------------------------------------------------------------------
extern "C" void kernel_launch(void* const* d_in, const int* in_sizes, int n_in,
                              void* d_out, int out_size, void* d_ws, size_t ws_size,
                              hipStream_t stream) {
  (void)in_sizes; (void)n_in; (void)out_size;
  const float* x    = (const float*)d_in[0];
  const float* mask = (const float*)d_in[1];
  const float* Wih0 = (const float*)d_in[2];
  const float* Whh0 = (const float*)d_in[3];
  const float* bih0 = (const float*)d_in[4];
  const float* bhh0 = (const float*)d_in[5];
  const float* Wih1 = (const float*)d_in[6];
  const float* Whh1 = (const float*)d_in[7];
  const float* bih1 = (const float*)d_in[8];
  const float* bhh1 = (const float*)d_in[9];

  char* ws = (char*)d_ws;
  __bf16*  pw0   = (__bf16*)(ws + OFF_PW0);
  __bf16*  pw1   = (__bf16*)(ws + OFF_PW1);
  __bf16*  pw2   = (__bf16*)(ws + OFF_PW2);
  __bf16*  pw3   = (__bf16*)(ws + OFF_PW3);
  float*   bias0 = (float*)(ws + OFF_BIAS0);
  float*   bias1 = (float*)(ws + OFF_BIAS1);
  __bf16*  h0    = (__bf16*)(ws + OFF_H0);
  __bf16*  h1    = (__bf16*)(ws + OFF_H1);
  unsigned* cnt  = (unsigned*)(ws + OFF_CNT);
  __bf16*  xbf   = (__bf16*)(ws + OFF_XBF);

  // Zero hidden-state buffers + sync counter (graph-capturable memset node).
  hipMemsetAsync(ws + OFF_H0, 0, ZERO_BYTES, stream);

  const int packBlocks = (H * H) / 256;   // 4096
  pack_weight<<<packBlocks, 256, 0, stream>>>(Wih0, pw0);
  pack_weight<<<packBlocks, 256, 0, stream>>>(Whh0, pw1);
  pack_weight<<<packBlocks, 256, 0, stream>>>(Wih1, pw2);
  pack_weight<<<packBlocks, 256, 0, stream>>>(Whh1, pw3);
  bias_sum<<<4, 256, 0, stream>>>(bih0, bhh0, bih1, bhh1, bias0, bias1);

  const bool xpre = (ws_size >= WS_NEED_XPRE);
  if (xpre) {
    const int sxBlocks = (B * T * H) / (256 * 8);  // 16384
    scale_x<<<sxBlocks, 256, 0, stream>>>(x, mask, xbf);
    rnn_persistent<true><<<NBLOCKS, NTHREADS, 0, stream>>>(
        x, mask, xbf, pw0, pw1, pw2, pw3, bias0, bias1, h0, h1, cnt,
        (float*)d_out);
  } else {
    rnn_persistent<false><<<NBLOCKS, NTHREADS, 0, stream>>>(
        x, mask, xbf, pw0, pw1, pw2, pw3, bias0, bias1, h0, h1, cnt,
        (float*)d_out);
  }
}
